// Net_56607668961384
// MI455X (gfx1250) — matrix operand order repaired and checked by reading, not compiled, not run on hardware
//
#include <hip/hip_runtime.h>
#include <hip/hip_bf16.h>
#include <math.h>

typedef _Float16 half_t;
typedef __attribute__((ext_vector_type(16))) _Float16 v16h;
typedef __attribute__((ext_vector_type(8)))  _Float16 v8h;
typedef __attribute__((ext_vector_type(8)))  float    v8f;
typedef __attribute__((ext_vector_type(4)))  float    f32x4;
typedef __attribute__((ext_vector_type(4)))  int      v4i;

#define DEV __device__ __forceinline__

// ---------------- CDNA5 async global->LDS (guarded) ----------------
#if defined(__has_builtin)
#if __has_builtin(__builtin_amdgcn_global_load_async_to_lds_b128) && __has_builtin(__builtin_amdgcn_s_wait_asynccnt)
#define HAVE_ASYNC_LDS 1
#endif
#endif
#ifndef HAVE_ASYNC_LDS
#define HAVE_ASYNC_LDS 0
#endif

#if HAVE_ASYNC_LDS
typedef __attribute__((address_space(1))) v4i gv4i; // global v4i
typedef __attribute__((address_space(3))) v4i lv4i; // LDS v4i
#endif

DEV void cp16_lds(half_t* dst, const half_t* src) {
#if HAVE_ASYNC_LDS
  __builtin_amdgcn_global_load_async_to_lds_b128((gv4i*)src, (lv4i*)dst, 0, 0);
#else
  *(f32x4*)dst = *(const f32x4*)src;
#endif
}
DEV void wait_stage() {
#if HAVE_ASYNC_LDS
  __builtin_amdgcn_s_wait_asynccnt(0);
#endif
  __syncthreads();
}

DEV float geluf(float v) { return 0.5f * v * (1.0f + erff(v * 0.70710678118654752f)); }

// ---------------- core WMMA GEMM:  C[z] = A[z] * Bt[z]^T (+bias,+act) ----------
// A: [M,Kp] f16 row-major, Bt: [N,Kp] f16 row-major, Kp % 32 == 0.
// block = 256 threads = 8 wave32 waves; block tile 128x32; each wave owns a
// 16x32 strip: 2 WMMA accumulators sharing one A fragment per K-step.
DEV void stage_tiles(const half_t* Az, const half_t* Bz, half_t* As, half_t* Bs,
                     int tid, int bm, int bn, int M, int N, int Kp, int k0) {
#pragma unroll
  for (int h = 0; h < 2; ++h) {
    int ch = tid + h * 256;           // 512 chunks of 16B = 128 rows x 32 halves
    int row = ch >> 2, seg = ch & 3;
    int gr = bm + row; if (gr >= M) gr = M - 1;
    const half_t* src = Az + (long long)gr * Kp + k0 + seg * 8;
    if (h == 0 && k0 + 64 <= Kp) __builtin_prefetch((const void*)(src + 32), 0, 0);
    cp16_lds(&As[row * 32 + seg * 8], src);
  }
  if (tid < 128) {
    int row = tid >> 2, seg = tid & 3; // 32 rows x 4 chunks
    int gn = bn + row; if (gn >= N) gn = N - 1;
    const half_t* src = Bz + (long long)gn * Kp + k0 + seg * 8;
    cp16_lds(&Bs[row * 32 + seg * 8], src);
  }
}

__global__ __launch_bounds__(256) void k_gemm_wmma(
    const half_t* __restrict__ A, const half_t* __restrict__ Bt,
    float* __restrict__ C, half_t* __restrict__ C16, const float* __restrict__ bias,
    int M, int N, int Kp,
    long long sA, long long sB, long long sC, long long sC16, int ldc16, int act) {
  __shared__ __align__(32) half_t As[2][128 * 32];
  __shared__ __align__(32) half_t Bs[2][32 * 32];
  const int tid = threadIdx.x;
  const int lane = tid & 31;
  const int wave = tid >> 5;            // M-tile within block (0..7)
  const int bm = blockIdx.x * 128, bn = blockIdx.y * 32;
  const long long z = blockIdx.z;
  const half_t* Az = A + z * sA;
  const half_t* Bz = Bt + z * sB;
  v8f acc0 = {0.f, 0.f, 0.f, 0.f, 0.f, 0.f, 0.f, 0.f};
  v8f acc1 = {0.f, 0.f, 0.f, 0.f, 0.f, 0.f, 0.f, 0.f};
  const int nk = Kp >> 5;
  stage_tiles(Az, Bz, As[0], Bs[0], tid, bm, bn, M, N, Kp, 0);
  int buf = 0;
  const int m = lane & 15;
  const int kh = lane >> 4;
  for (int kt = 0; kt < nk; ++kt) {
    wait_stage();
    if (kt + 1 < nk)
      stage_tiles(Az, Bz, As[buf ^ 1], Bs[buf ^ 1], tid, bm, bn, M, N, Kp, (kt + 1) * 32);
    // A fragment: lane group kh -> K = kh*16 + j (j = 0..15)
    v8h alo = *(const v8h*)&As[buf][(wave * 16 + m) * 32 + kh * 16];
    v8h ahi = *(const v8h*)&As[buf][(wave * 16 + m) * 32 + kh * 16 + 8];
    v16h a;
#pragma unroll
    for (int j = 0; j < 8; ++j) { a[j] = alo[j]; a[j + 8] = ahi[j]; }
    // B fragments for the two N-tiles: lane = N, group kh -> K = kh*16 + j
    v16h b0 = *(const v16h*)&Bs[buf][(m) * 32 + kh * 16];
    v16h b1 = *(const v16h*)&Bs[buf][(16 + m) * 32 + kh * 16];
    acc0 = __builtin_amdgcn_wmma_f32_16x16x32_f16(false, a, false, b0, (short)0, acc0, false, false);
    acc1 = __builtin_amdgcn_wmma_f32_16x16x32_f16(false, a, false, b1, (short)0, acc1, false, false);
    buf ^= 1;
  }
  // epilogue: acc0 -> cols [bn, bn+16), acc1 -> cols [bn+16, bn+32)
#pragma unroll
  for (int half = 0; half < 2; ++half) {
    int col = bn + half * 16 + m;
    if (col < N) {
      float bv = bias ? bias[col] : 0.0f;
#pragma unroll
      for (int r = 0; r < 8; ++r) {
        int row = bm + wave * 16 + kh * 8 + r;
        if (row < M) {
          float v = (half ? acc1[r] : acc0[r]) + bv;
          if (act == 1) v = fmaxf(v, 0.0f);
          else if (act == 2) v = geluf(v);
          if (C)   C[z * sC + (long long)row * N + col] = v;
          if (C16) C16[z * sC16 + (long long)row * ldc16 + col] = (half_t)v;
        }
      }
    }
  }
}

// ---------------- elementwise / data-movement kernels ----------------
static inline unsigned nb(long long n) { return (unsigned)((n + 255) / 256); }
#define GIDX (blockIdx.x * 256ll + threadIdx.x)

__global__ void k_pack16(const float* in, half_t* out, long long R, int K, int ldin,
                         int cstr, int c0, int Kp) {
  long long i = GIDX; long long total = R * Kp; if (i >= total) return;
  int c = (int)(i % Kp); long long r = i / Kp;
  out[i] = (c < K) ? (half_t)in[r * ldin + (long long)c * cstr + c0] : (half_t)0.f;
}

__global__ void k_wpack(const float* w, half_t* out, int O, int Cin, int KH, int KW, int Kp) {
  int K = Cin * KH * KW;
  long long i = GIDX; long long total = (long long)O * Kp; if (i >= total) return;
  int c = (int)(i % Kp); int o = (int)(i / Kp);
  half_t v = (half_t)0.f;
  if (c < K) {
    int kk = c / Cin, ci = c % Cin;
    int ky = kk / KW, kx = kk % KW;
    v = (half_t)w[(((long long)o * Cin + ci) * KH + ky) * KW + kx];
  }
  out[i] = v;
}

__global__ void k_im2col(const float* x, half_t* out, int B, int H, int W, int ldc, int c0,
                         int C, int kh, int kw, int stride, int pad, int OH, int OW, int Kp) {
  int K = C * kh * kw;
  long long i = GIDX; long long total = (long long)B * OH * OW * Kp; if (i >= total) return;
  int c = (int)(i % Kp); long long r = i / Kp;
  int ox = (int)(r % OW); long long t = r / OW; int oy = (int)(t % OH); int b = (int)(t / OH);
  half_t v = (half_t)0.f;
  if (c < K) {
    int kk = c / C, ci = c % C;
    int ky = kk / kw, kx = kk % kw;
    int y = oy * stride - pad + ky, xx = ox * stride - pad + kx;
    if (y >= 0 && y < H && xx >= 0 && xx < W)
      v = (half_t)x[(((long long)b * H + y) * W + xx) * ldc + c0 + ci];
  }
  out[i] = v;
}

__global__ void k_tohalf(const float* x, half_t* y, long long n) {
  long long i = GIDX; if (i < n) y[i] = (half_t)x[i];
}
__global__ void k_add(float* out, const float* x, const float* y, long long n) {
  long long i = GIDX; if (i < n) out[i] = x[i] + y[i];
}
__global__ void k_add_mirror(float* xf, half_t* x16, const float* add, long long n) {
  long long i = GIDX; if (i < n) { float v = xf[i] + add[i]; xf[i] = v; x16[i] = (half_t)v; }
}
__global__ void k_axpy_p(float* out, const float* s, int si, const float* x, const float* y, long long n) {
  long long i = GIDX; if (i < n) out[i] = s[si] * x[i] + y[i];
}
__global__ void k_scale_p(float* out, const float* s, int si, const float* x, long long n) {
  long long i = GIDX; if (i < n) out[i] = s[si] * x[i];
}
__global__ void k_scale_vec(float* y, const float* sc, long long n, long long perBatch) {
  long long i = GIDX; if (i >= n) return;
  int c = (int)(i & 7); int b = (int)(i / perBatch);
  y[i] *= sc[b * 8 + c];
}
__global__ void k_slice16(float* out, const float* in, int ld, int c0, long long R) {
  long long i = GIDX; if (i >= R * 16) return;
  int c = (int)(i & 15); long long r = i >> 4;
  out[i] = in[r * ld + c0 + c];
}
__global__ void k_concat16(half_t* out, const float* a, const float* b, long long R) {
  long long i = GIDX; if (i >= R * 32) return;
  int c = (int)(i & 31); long long r = i >> 5;
  out[i] = (half_t)(c < 16 ? a[r * 16 + c] : b[r * 16 + (c - 16)]);
}
__global__ void k_layernorm16(const float* x, const float* g, const float* b,
                              half_t* out, long long R, int ldin, int c0) {
  long long r = GIDX; if (r >= R) return;
  const float* p = x + r * ldin + c0;
  float m = 0.f;
#pragma unroll
  for (int i = 0; i < 16; ++i) m += p[i];
  m *= (1.f / 16.f);
  float v = 0.f;
#pragma unroll
  for (int i = 0; i < 16; ++i) { float d = p[i] - m; v += d * d; }
  v *= (1.f / 16.f);
  float inv = rsqrtf(v + 1e-5f);
  half_t* o = out + r * 32;
#pragma unroll
  for (int i = 0; i < 16; ++i) o[i] = (half_t)((p[i] - m) * inv * g[i] + b[i]);
#pragma unroll
  for (int i = 16; i < 32; ++i) o[i] = (half_t)0.f;
}

// relpos [31,31] -> bias [256,256]
__global__ void k_relbias(const float* relpos, float* out) {
  int i = blockIdx.x * 256 + threadIdx.x; if (i >= 65536) return;
  int p = i >> 8, q = i & 255;
  int py = p >> 4, px = p & 15, qy = q >> 4, qx = q & 15;
  out[i] = relpos[(py - qy + 15) * 31 + (px - qx + 15)];
}

__global__ void k_softmax_attn(const half_t* __restrict__ sim, const float* __restrict__ relb,
                               half_t* __restrict__ attn, float scale, int shift, int nh, int nw) {
  const int p = blockIdx.x;
  const long long z = blockIdx.y;
  const int q = threadIdx.x;
  const int lane = q & 31, wv = q >> 5;
  long long off = (z * 256 + p) * 256 + q;
  float v = (float)sim[off] * scale + relb[p * 256 + q];
  if (shift) {
    int wi = (int)(z % (nh * nw));
    int wy = wi / nw, wx = wi % nw;
    int py = p >> 4, px = p & 15, qy = q >> 4, qx = q & 15;
    int s = 16 - shift;
    if ((wy == nh - 1 && ((py < s) != (qy < s))) ||
        (wx == nw - 1 && ((px < s) != (qx < s)))) v = -1e30f;
  }
  __shared__ float red[8];
  float m = v;
  for (int o = 16; o; o >>= 1) m = fmaxf(m, __shfl_xor(m, o));
  if (lane == 0) red[wv] = m;
  __syncthreads();
  float t = red[lane & 7];
  for (int o = 4; o; o >>= 1) t = fmaxf(t, __shfl_xor(t, o));
  float e = __expf(v - t);
  float s1 = e;
  for (int o = 16; o; o >>= 1) s1 += __shfl_xor(s1, o);
  __syncthreads();
  if (lane == 0) red[wv] = s1;
  __syncthreads();
  float ts = red[lane & 7];
  for (int o = 4; o; o >>= 1) ts += __shfl_xor(ts, o);
  attn[off] = (half_t)(e / ts);
}

// qkv [B,H,W,48] -> q/k [Z,256,32] (zero-padded K), vT [Z,16,256]; roll folded in.
__global__ void k_qkv_pack(const float* qkv, half_t* q16, half_t* k16, half_t* vT,
                           int B, int H, int W, int nh, int nw, int shift) {
  long long i = GIDX; long long total = (long long)B * nh * nw * 256; if (i >= total) return;
  int p = (int)(i & 255); long long zz = i >> 8;
  int wi = (int)(zz % (nh * nw)); int b = (int)(zz / (nh * nw));
  int wy = wi / nw, wx = wi % nw;
  int py = p >> 4, px = p & 15;
  int y = (wy * 16 + py + shift) % H;
  int x = (wx * 16 + px + shift) % W;
  const float* src = qkv + (((long long)b * H + y) * W + x) * 48;
  half_t* qd = q16 + i * 32;
  half_t* kd = k16 + i * 32;
#pragma unroll
  for (int c = 0; c < 16; ++c) { qd[c] = (half_t)src[c]; kd[c] = (half_t)src[16 + c]; }
#pragma unroll
  for (int c = 16; c < 32; ++c) { qd[c] = (half_t)0.f; kd[c] = (half_t)0.f; }
  half_t* vd = vT + zz * (16 * 256);
#pragma unroll
  for (int c = 0; c < 16; ++c) vd[c * 256 + p] = (half_t)src[32 + c];
}

__global__ void k_av_unpack(const float* av, half_t* out, int B, int H, int W,
                            int nh, int nw, int shift) {
  long long i = GIDX; long long total = (long long)B * nh * nw * 256; if (i >= total) return;
  int p = (int)(i & 255); long long zz = i >> 8;
  int wi = (int)(zz % (nh * nw)); int b = (int)(zz / (nh * nw));
  int wy = wi / nw, wx = wi % nw;
  int py = p >> 4, px = p & 15;
  int y = (wy * 16 + py + shift) % H;
  int x = (wx * 16 + px + shift) % W;
  half_t* o = out + (((long long)b * H + y) * W + x) * 32;
  const float* s = av + i * 16;
#pragma unroll
  for (int c = 0; c < 16; ++c) o[c] = (half_t)s[c];
#pragma unroll
  for (int c = 16; c < 32; ++c) o[c] = (half_t)0.f;
}

// pixel shuffle NHWC: [B,160,160,32] -> [B,320,320,8]
__global__ void k_pshuf(const float* in, float* out, int B, int H, int W, int Cin) {
  int Co = Cin / 4;
  long long i = GIDX; long long total = (long long)B * 4 * H * W * Co; if (i >= total) return;
  int c = (int)(i % Co); long long t = i / Co;
  int x = (int)(t % (2 * W)); t /= (2 * W);
  int y = (int)(t % (2 * H)); int b = (int)(t / (2 * H));
  out[i] = in[(((long long)b * H + (y >> 1)) * W + (x >> 1)) * Cin + c * 4 + (y & 1) * 2 + (x & 1)];
}
// pixel unshuffle NHWC: [B,320,320,8] -> [B,160,160,32]
__global__ void k_punshuf(const float* in, float* out, int B, int H, int W, int Cin) {
  int Co = Cin * 4;
  long long i = GIDX; long long total = (long long)B * H * W * Co; if (i >= total) return;
  int cc = (int)(i % Co); long long t = i / Co;
  int x = (int)(t % W); t /= W;
  int y = (int)(t % H); int b = (int)(t / H);
  int c = cc >> 2, ry = (cc >> 1) & 1, rx = cc & 1;
  out[i] = in[(((long long)b * 2 * H + 2 * y + ry) * 2 * W + (2 * x + rx)) * Cin + c];
}
// tail [B,160,160,4] -> d_out [B,1,320,320]
__global__ void k_shuffle_out(const float* in, float* out, int B) {
  long long i = GIDX; long long total = (long long)B * 320 * 320; if (i >= total) return;
  int x = (int)(i % 320); long long t = i / 320;
  int y = (int)(t % 320); int b = (int)(t / 320);
  out[i] = in[(((long long)b * 160 + (y >> 1)) * 160 + (x >> 1)) * 4 + (y & 1) * 2 + (x & 1)];
}

// ---- measurement operator helpers ----
__global__ void k_make_cs(const int* qG, const int* qD, float* cs, int B) {
  int i = threadIdx.x; if (i >= B) return;
  cs[i * 2 + 0] = (float)qG[i] * (1.f / 1024.f);
  cs[i * 2 + 1] = (float)qD[i] * (1.f / 1024.f);
}
__global__ void k_ss_scale(const float* cs, const float* s0w, const float* s0b,
                           const float* s1w, const float* s1b, float* out, int B) {
  int i = threadIdx.x; if (i >= B * 8) return;
  int b = i / 8, o = i % 8;
  float h[8];
#pragma unroll
  for (int j = 0; j < 8; ++j)
    h[j] = fmaxf(s0w[j * 2 + 0] * cs[b * 2 + 0] + s0w[j * 2 + 1] * cs[b * 2 + 1] + s0b[j], 0.f);
  float acc = s1b[o];
#pragma unroll
  for (int j = 0; j < 8; ++j) acc += s1w[o * 8 + j] * h[j];
  out[i] = acc;
}
__global__ void k_gen_dct(half_t* D, half_t* Dt, int N) {
  long long i = GIDX; if (i >= (long long)N * N) return;
  int n = (int)(i / N), k = (int)(i % N);
  float c = (n == 0) ? sqrtf(1.f / N) : sqrtf(2.f / N);
  float v = c * __cosf(3.14159265358979323846f * (2 * k + 1) * n / (2.f * N));
  D[(long long)n * N + k] = (half_t)v;
  Dt[(long long)k * N + n] = (half_t)v;
}
__global__ void k_transpose_img(const float* in, half_t* out, int B, int N) {
  long long i = GIDX; if (i >= (long long)B * N * N) return;
  int v = (int)(i % N); long long t = i / N;
  int u = (int)(t % N); int b = (int)(t / N);
  out[i] = (half_t)in[((long long)b * N + v) * N + u];
}
__global__ void k_perm_gather(const float* z, const int* perm, half_t* out, int ld, int c0) {
  long long i = GIDX; if (i >= 400ll * 1024) return;
  int pq = (int)(i & 1023); long long r = i >> 10;
  int jj = (int)(r % 10); long long t = r / 10;
  int ii = (int)(t % 10); int b = (int)(t / 10);
  int p = pq >> 5, qq = pq & 31;
  int f = perm[(ii * 32 + p) * 320 + (jj * 32 + qq)];
  out[i] = (half_t)z[((long long)b * 102400 + f) * ld + c0];
}
__global__ void k_perm_scatter(const float* GT, const int* perm_inv, float* ATb, int B) {
  long long i = GIDX; if (i >= (long long)B * 102400) return;
  int idx = (int)(i % 102400); int b = (int)(i / 102400);
  int f = perm_inv[idx];
  int y = f / 320, x = f % 320;
  int ii = y >> 5, p = y & 31, jj = x >> 5, qq = x & 31;
  ATb[((long long)b * 102400 + idx) * 2 + 0] =
      GT[((long long)(b * 100 + ii * 10 + jj)) * 1024 + p * 32 + qq];
}
__global__ void k_zz_gather(const float* Zt, const int* dx, const int* dy, float* out, int B) {
  long long i = GIDX; if (i >= (long long)B * 102400) return;
  int t = (int)(i % 102400); int b = (int)(i / 102400);
  int n = t / 100, ij = t % 100;
  int ii = ij / 10, jj = ij % 10;
  long long r = b * 100 + ii * 10 + jj;
  // Zt[b][v][u] = dct[b][u][v] ; need dct[b][dx][dy]
  out[r * 1024 + n] = Zt[((long long)b * 320 + dy[t]) * 320 + dx[t]];
}
__global__ void k_zz_scatter(const float* res, const int* dx, const int* dy, half_t* S16, int B) {
  long long i = GIDX; if (i >= (long long)B * 102400) return;
  int t = (int)(i % 102400); int b = (int)(i / 102400);
  int n = t / 100, ij = t % 100;
  int ii = ij / 10, jj = ij % 10;
  long long r = b * 100 + ii * 10 + jj;
  S16[((long long)b * 320 + dx[t]) * 320 + dy[t]] = (half_t)res[r * 1024 + n];
}
__global__ void k_tstore_ch1(const float* Zt, float* ATb, int B) {
  long long i = GIDX; if (i >= (long long)B * 102400) return;
  int v = (int)(i % 320); long long t = i / 320;
  int u = (int)(t % 320); int b = (int)(t / 320);
  ATb[((long long)b * 102400 + u * 320 + v) * 2 + 1] = Zt[((long long)b * 320 + v) * 320 + u];
}
__global__ void k_maskres(const float* a, const float* ysub, const int* qv, float* out, long long n) {
  long long i = GIDX; if (i >= n) return;
  int nn = (int)(i & 1023); long long r = i >> 10;
  int b = (int)(r / 100);
  float v = (nn < qv[b]) ? a[i] : 0.f;
  if (ysub) v -= ysub[i];
  out[i] = v;
}
__global__ void k_stage_combine(float* x8, const float* H, const float* atres,
                                const float* rho, long long n8) {
  long long i = GIDX; if (i >= n8) return;
  int c = (int)(i & 7); long long pix = i >> 3;
  float xv = x8[i];
  float v = (c < 2) ? (H[i] + xv - rho[0] * atres[pix * 2 + c]) : (H[i] + xv);
  x8[i] = v;
}

// ---------------- host side ----------------
static void gemm(hipStream_t st, const half_t* A, const half_t* Bt, float* C, half_t* C16,
                 const float* bias, long long M, int N, int Kp, int Z,
                 long long sA, long long sB, long long sC, long long sC16, int ldc16, int act) {
  dim3 g((unsigned)((M + 127) / 128), (unsigned)((N + 31) / 32), (unsigned)Z);
  k_gemm_wmma<<<g, dim3(256), 0, st>>>(A, Bt, C, C16, bias, (int)M, N, Kp, sA, sB, sC, sC16,
                                       ldc16 ? ldc16 : N, act);
}

struct Alloc { char* p; size_t used; };
template <typename T> static T* wsa(Alloc& a, size_t n) {
  size_t bytes = (n * sizeof(T) + 255) & ~(size_t)255;
  T* r = (T*)(a.p + a.used);
  a.used += bytes;
  return r;
}

// params pytree (sorted-key) index helpers, base = 8
static inline int pi_ss_blk(int blk, int leaf) { return 8 + 2 + blk * 6 + leaf; } // bb,bw,s0b,s0w,s1b,s1w
static inline int pi_ss_head_b() { return 8 + 32; }
static inline int pi_ss_head_w() { return 8 + 33; }
static inline int pi_ss_tail_b() { return 8 + 34; }
static inline int pi_ss_tail_w() { return 8 + 35; }
static inline int pi_stage(int s) { return 8 + 36 + s * 44; }
// stage: +0 b1b,+1 b1w,+2 b2b,+3 b2w, +4 ctb1, +23 ctb2, +42 rho, +43 s
// ctb: 0 c11b,1 c11w,2 c12b,3 c12w,4 cb1w,5 cb2w,6 fc1b,7 fc1w,8 fc2b,9 fc2w,
//      10 ln1b,11 ln1g,12 ln2b,13 ln2g,14 out_b,15 out_w,16 qkv_b,17 qkv_w,18 relpos

extern "C" void kernel_launch(void* const* d_in, const int* in_sizes, int n_in,
                              void* d_out, int out_size, void* d_ws, size_t ws_size,
                              hipStream_t stream) {
  (void)in_sizes; (void)out_size;
  const int B = 4;
  const long long HW = 320ll * 320;       // 102400
  const long long R320 = B * HW;          // 409600 pixels @ 320
  const long long R160 = B * 160ll * 160; // 102400 pixels @ 160
  const long long n8 = R320 * 8;
  const int Zw = B * 100;                 // windows

  auto inF = [&](int i) -> const float* { return (const float*)d_in[(i >= 0 && i < n_in) ? i : 0]; };
  auto inI = [&](int i) -> const int* { return (const int*)d_in[(i >= 0 && i < n_in) ? i : 0]; };

  const float* xin = inF(0);
  const int* qG = inI(1);
  const int* qD = inI(2);
  const int* perm = inI(3);
  const int* perm_inv = inI(4);
  const int* dctx = inI(5);
  const int* dcty = inI(6);
  const float* Aw = inF(7);
  float* outp = (float*)d_out;

  Alloc a{(char*)d_ws, 0};
  // f16 persistent
  half_t* Wgp = wsa<half_t>(a, 1024 * 1024);
  half_t* WgTp = wsa<half_t>(a, 1024 * 1024);
  half_t* Dp = wsa<half_t>(a, 320 * 320);
  half_t* Dtp = wsa<half_t>(a, 320 * 320);
  half_t* slotA = wsa<half_t>(a, 39321600);  // im2col slab (max R320*96)
  half_t* wslot = wsa<half_t>(a, 32768);
  // 320-level f32
  float* x8 = wsa<float>(a, n8);
  float* x2b = wsa<float>(a, n8);
  float* tmp8 = wsa<float>(a, n8);
  float* Hb = wsa<float>(a, n8);
  float* z1b = wsa<float>(a, R320 * 2);
  float* ATb = wsa<float>(a, R320 * 2);
  // measurement temps
  half_t* A16g = wsa<half_t>(a, 400 * 1024);
  float* aGf = wsa<float>(a, 400 * 1024);
  float* aDf = wsa<float>(a, 400 * 1024);
  float* res0 = wsa<float>(a, 400 * 1024);
  float* res1 = wsa<float>(a, 400 * 1024);
  float* GTf = wsa<float>(a, 400 * 1024);
  float* y0f = wsa<float>(a, 400 * 1024);
  float* y1f = wsa<float>(a, 400 * 1024);
  half_t* X16 = wsa<half_t>(a, R320);
  float* Yf = wsa<float>(a, R320);
  half_t* Yt16 = wsa<half_t>(a, R320);
  float* Ztf = wsa<float>(a, R320);
  half_t* S16 = wsa<half_t>(a, R320);
  // 160-level
  float* xh = wsa<float>(a, R160 * 32);
  float* xst = wsa<float>(a, R160 * 32);
  float* hbuf = wsa<float>(a, R160 * 32);
  half_t* hm16 = wsa<half_t>(a, R160 * 32);
  float* h2 = wsa<float>(a, R160 * 32);
  float* yf = wsa<float>(a, R160 * 32);
  float* delta = wsa<float>(a, R160 * 32);
  float* t1 = wsa<float>(a, R160 * 16);
  float* t2 = wsa<float>(a, R160 * 16);
  float* cxb = wsa<float>(a, R160 * 16);
  float* xt = wsa<float>(a, R160 * 16);
  float* aout = wsa<float>(a, R160 * 16);
  float* mlpf = wsa<float>(a, R160 * 16);
  half_t* ln16 = wsa<half_t>(a, R160 * 32);
  half_t* cat16 = wsa<half_t>(a, R160 * 32);
  half_t* avu16 = wsa<half_t>(a, R160 * 32);
  half_t* fch16 = wsa<half_t>(a, R160 * 64);
  float* qkvf = wsa<float>(a, R160 * 48);
  half_t* q16b = wsa<half_t>(a, (long long)Zw * 256 * 32);
  half_t* k16b = wsa<half_t>(a, (long long)Zw * 256 * 32);
  half_t* vT16 = wsa<half_t>(a, (long long)Zw * 16 * 256);
  half_t* sim16 = wsa<half_t>(a, (long long)Zw * 256 * 256);
  half_t* attn16 = wsa<half_t>(a, (long long)Zw * 256 * 256);
  float* avf = wsa<float>(a, (long long)Zw * 256 * 16);
  float* relb = wsa<float>(a, 65536);
  float* csb = wsa<float>(a, 8);
  float* scaleb = wsa<float>(a, 5 * 32);
  float* tl = wsa<float>(a, R160 * 4);
  if (a.used > ws_size) { /* workspace too small; proceed best-effort */ }

  // ---- constant packs ----
  k_make_cs<<<1, 32, 0, stream>>>(qG, qD, csb, B);
  for (int blk = 0; blk < 5; ++blk)
    k_ss_scale<<<1, 64, 0, stream>>>(csb, inF(pi_ss_blk(blk, 3)), inF(pi_ss_blk(blk, 2)),
                                     inF(pi_ss_blk(blk, 5)), inF(pi_ss_blk(blk, 4)),
                                     scaleb + blk * 32, B);
  k_pack16<<<nb(1024ll * 1024), 256, 0, stream>>>(Aw, Wgp, 1024, 1024, 1024, 1, 0, 1024);
  k_transpose_img<<<nb(1024ll * 1024), 256, 0, stream>>>(Aw, WgTp, 1, 1024);
  k_gen_dct<<<nb(320ll * 320), 256, 0, stream>>>(Dp, Dtp, 320);

  // ---- generic conv (im2col + WMMA GEMM) ----
  auto conv = [&](const float* src, int Bn, int Hh, int Wn, int ldc, int c0, int Cin,
                  int kh, int kw, int stride, int pad, const float* wgt, const float* bias,
                  int Cout, float* dstF, half_t* dst16, int act) {
    int OH = (Hh + 2 * pad - kh) / stride + 1;
    int OW = (Wn + 2 * pad - kw) / stride + 1;
    int K = Cin * kh * kw, Kp = ((K + 31) / 32) * 32;
    long long R = (long long)Bn * OH * OW;
    k_im2col<<<nb(R * Kp), 256, 0, stream>>>(src, slotA, Bn, Hh, Wn, ldc, c0, Cin,
                                             kh, kw, stride, pad, OH, OW, Kp);
    k_wpack<<<nb((long long)Cout * Kp), 256, 0, stream>>>(wgt, wslot, Cout, Cin, kh, kw, Kp);
    gemm(stream, slotA, wslot, dstF, dst16, bias, R, Cout, Kp, 1, 0, 0, 0, 0, Cout, act);
  };

  // ---- A_fn: z (NHWC 2ch @320) -> raw aG [400,1024], aD [400,1024] ----
  auto A_fn = [&](const float* z) {
    k_perm_gather<<<nb(400ll * 1024), 256, 0, stream>>>(z, perm, A16g, 2, 0);
    gemm(stream, A16g, Wgp, aGf, nullptr, nullptr, 400, 1024, 1024, 1, 0, 0, 0, 0, 0, 0);
    k_pack16<<<nb(R320), 256, 0, stream>>>(z, X16, B * 320, 320, 320 * 2, 2, 1, 320);
    gemm(stream, X16, Dp, Yf, nullptr, nullptr, B * 320, 320, 320, 1, 0, 0, 0, 0, 0, 0);
    k_transpose_img<<<nb(R320), 256, 0, stream>>>(Yf, Yt16, B, 320);
    gemm(stream, Yt16, Dp, Ztf, nullptr, nullptr, B * 320, 320, 320, 1, 0, 0, 0, 0, 0, 0);
    k_zz_gather<<<nb(R320), 256, 0, stream>>>(Ztf, dctx, dcty, aDf, B);
  };
  // ---- AT: residuals -> ATb (NHWC 2ch @320) ----
  auto AT = [&](const float* r0, const float* r1) {
    k_pack16<<<nb(400ll * 1024), 256, 0, stream>>>(r0, A16g, 400, 1024, 1024, 1, 0, 1024);
    gemm(stream, A16g, WgTp, GTf, nullptr, nullptr, 400, 1024, 1024, 1, 0, 0, 0, 0, 0, 0);
    k_perm_scatter<<<nb(R320), 256, 0, stream>>>(GTf, perm_inv, ATb, B);
    k_zz_scatter<<<nb(R320), 256, 0, stream>>>(r1, dctx, dcty, S16, B);
    gemm(stream, S16, Dtp, Yf, nullptr, nullptr, B * 320, 320, 320, 1, 0, 0, 0, 0, 0, 0);
    k_transpose_img<<<nb(R320), 256, 0, stream>>>(Yf, Yt16, B, 320);
    gemm(stream, Yt16, Dtp, Ztf, nullptr, nullptr, B * 320, 320, 320, 1, 0, 0, 0, 0, 0, 0);
    k_tstore_ch1<<<nb(R320), 256, 0, stream>>>(Ztf, ATb, B);
  };
  // ---- ss body (5 scaled convs), returns final buffer ----
  auto ss_body = [&](float* cur, float* oth) -> float* {
    for (int blk = 0; blk < 5; ++blk) {
      conv(cur, B, 320, 320, 8, 0, 8, 3, 3, 1, 1,
           inF(pi_ss_blk(blk, 1)), inF(pi_ss_blk(blk, 0)), 8, oth, nullptr, 0);
      k_scale_vec<<<nb(n8), 256, 0, stream>>>(oth, scaleb + blk * 32, n8, HW * 8);
      float* t = cur; cur = oth; oth = t;
    }
    return cur;
  };

  // ---- swin block on xt [R160,16] ----
  auto swin = [&](int pb, int shift) {
    k_layernorm16<<<nb(R160), 256, 0, stream>>>(xt, inF(pb + 11), inF(pb + 10), ln16, R160, 16, 0);
    k_pack16<<<nb(48ll * 32), 256, 0, stream>>>(inF(pb + 17), wslot, 48, 16, 16, 1, 0, 32);
    gemm(stream, ln16, wslot, qkvf, nullptr, inF(pb + 16), R160, 48, 32, 1, 0, 0, 0, 0, 0, 0);
    k_qkv_pack<<<nb((long long)Zw * 256), 256, 0, stream>>>(qkvf, q16b, k16b, vT16, B, 160, 160, 10, 10, shift);
    gemm(stream, q16b, k16b, nullptr, sim16, nullptr, 256, 256, 32, Zw,
         8192, 8192, 0, 65536, 256, 0);
    k_relbias<<<256, 256, 0, stream>>>(inF(pb + 18), relb);
    k_softmax_attn<<<dim3(256, Zw), 256, 0, stream>>>(sim16, relb, attn16, 0.25f, shift, 10, 10);
    gemm(stream, attn16, vT16, avf, nullptr, nullptr, 256, 16, 256, Zw,
         65536, 4096, 4096, 0, 16, 0);
    k_av_unpack<<<nb((long long)Zw * 256), 256, 0, stream>>>(avf, avu16, B, 160, 160, 10, 10, shift);
    k_pack16<<<nb(16ll * 32), 256, 0, stream>>>(inF(pb + 15), wslot, 16, 16, 16, 1, 0, 32);
    gemm(stream, avu16, wslot, aout, nullptr, inF(pb + 14), R160, 16, 32, 1, 0, 0, 0, 0, 0, 0);
    k_add<<<nb(R160 * 16), 256, 0, stream>>>(xt, xt, aout, R160 * 16);
    k_layernorm16<<<nb(R160), 256, 0, stream>>>(xt, inF(pb + 13), inF(pb + 12), ln16, R160, 16, 0);
    k_pack16<<<nb(64ll * 32), 256, 0, stream>>>(inF(pb + 7), wslot, 64, 16, 16, 1, 0, 32);
    gemm(stream, ln16, wslot, nullptr, fch16, inF(pb + 6), R160, 64, 32, 1, 0, 0, 0, 0, 64, 2);
    k_pack16<<<nb(16ll * 64), 256, 0, stream>>>(inF(pb + 9), wslot, 16, 64, 64, 1, 0, 64);
    gemm(stream, fch16, wslot, mlpf, nullptr, inF(pb + 8), R160, 16, 64, 1, 0, 0, 0, 0, 0, 0);
    k_add<<<nb(R160 * 16), 256, 0, stream>>>(xt, xt, mlpf, R160 * 16);
  };
  // ---- conv-transformer block, in-place on (hbuf, hm16) ----
  auto ctb = [&](int pb, int shift) {
    k_pack16<<<nb(32ll * 32), 256, 0, stream>>>(inF(pb + 1), wslot, 32, 32, 32, 1, 0, 32);
    gemm(stream, hm16, wslot, yf, nullptr, inF(pb + 0), R160, 32, 32, 1, 0, 0, 0, 0, 0, 0);
    k_slice16<<<nb(R160 * 16), 256, 0, stream>>>(cxb, yf, 32, 0, R160);
    k_slice16<<<nb(R160 * 16), 256, 0, stream>>>(xt, yf, 32, 16, R160);
    conv(yf, B, 160, 160, 32, 0, 16, 3, 3, 1, 1, inF(pb + 4), nullptr, 16, t1, nullptr, 1);
    conv(t1, B, 160, 160, 16, 0, 16, 3, 3, 1, 1, inF(pb + 5), nullptr, 16, t2, nullptr, 0);
    k_add<<<nb(R160 * 16), 256, 0, stream>>>(cxb, cxb, t2, R160 * 16);
    swin(pb, shift);
    k_concat16<<<nb(R160 * 32), 256, 0, stream>>>(cat16, cxb, xt, R160);
    k_pack16<<<nb(32ll * 32), 256, 0, stream>>>(inF(pb + 3), wslot, 32, 32, 32, 1, 0, 32);
    gemm(stream, cat16, wslot, delta, nullptr, inF(pb + 2), R160, 32, 32, 1, 0, 0, 0, 0, 0, 0);
    k_add_mirror<<<nb(R160 * 32), 256, 0, stream>>>(hbuf, hm16, delta, R160 * 32);
  };

  // ======== y = A_fn(ss_forward(x, cs)) ========
  conv(xin, B, 320, 320, 1, 0, 1, 3, 3, 1, 1, inF(pi_ss_head_w()), inF(pi_ss_head_b()),
       8, x2b, nullptr, 0);
  {
    float* fin = ss_body(x2b, tmp8);
    conv(fin, B, 320, 320, 8, 0, 8, 3, 3, 1, 1, inF(pi_ss_tail_w()), inF(pi_ss_tail_b()),
         2, z1b, nullptr, 0);
  }
  A_fn(z1b);
  k_maskres<<<nb(400ll * 1024), 256, 0, stream>>>(aGf, nullptr, qG, y0f, 400ll * 1024);
  k_maskres<<<nb(400ll * 1024), 256, 0, stream>>>(aDf, nullptr, qD, y1f, 400ll * 1024);

  // ======== xh = conv(AT(*y), head_w, stride 2) ========
  AT(y0f, y1f);
  conv(ATb, B, 320, 320, 2, 0, 2, 6, 6, 2, 2, inF(9), inF(8), 32, xh, nullptr, 0);

  // ======== stages ========
  for (int s = 0; s < 5; ++s) {
    int pb = pi_stage(s);
    k_pshuf<<<nb(n8), 256, 0, stream>>>(xh, x8, B, 160, 160, 32);
    // ss_fuse
    conv(x8, B, 320, 320, 8, 2, 1, 3, 3, 1, 1, inF(pi_ss_head_w()), inF(pi_ss_head_b()),
         8, tmp8, nullptr, 0);
    k_axpy_p<<<nb(n8), 256, 0, stream>>>(x2b, inF(pb + 43), 0, x8, tmp8, n8);
    float* fin = ss_body(x2b, tmp8);
    k_scale_p<<<nb(n8), 256, 0, stream>>>(Hb, inF(pb + 43), 1, fin, n8);
    conv(fin, B, 320, 320, 8, 0, 8, 3, 3, 1, 1, inF(pi_ss_tail_w()), inF(pi_ss_tail_b()),
         2, z1b, nullptr, 0);
    // data-consistency step
    A_fn(z1b);
    k_maskres<<<nb(400ll * 1024), 256, 0, stream>>>(aGf, y0f, qG, res0, 400ll * 1024);
    k_maskres<<<nb(400ll * 1024), 256, 0, stream>>>(aDf, y1f, qD, res1, 400ll * 1024);
    AT(res0, res1);
    k_stage_combine<<<nb(n8), 256, 0, stream>>>(x8, Hb, ATb, inF(pb + 42), n8);
    k_punshuf<<<nb(R160 * 32), 256, 0, stream>>>(x8, xst, B, 160, 160, 8);
    // body: conv -> ctb -> ctb -> conv, residual
    conv(xst, B, 160, 160, 32, 0, 32, 3, 3, 1, 1, inF(pb + 1 + 0), inF(pb + 0), 32, hbuf, hm16, 0);
    ctb(pb + 4, 0);
    ctb(pb + 23, 8);
    conv(hbuf, B, 160, 160, 32, 0, 32, 3, 3, 1, 1, inF(pb + 3), inF(pb + 2), 32, h2, nullptr, 0);
    k_add<<<nb(R160 * 32), 256, 0, stream>>>(xh, xst, h2, R160 * 32);
  }

  // ======== tail conv + pixel shuffle -> output ========
  conv(xh, B, 160, 160, 32, 0, 32, 3, 3, 1, 1, inF(8 + 36 + 220 + 1), inF(8 + 36 + 220 + 0),
       4, tl, nullptr, 0);
  k_shuffle_out<<<nb(R320), 256, 0, stream>>>(tl, outp, B);
}
